// Net_57243324121376
// MI455X (gfx1250) — compile-verified, matrix operand order checked by reading, not run on hardware
//
#include <hip/hip_runtime.h>
#include <hip/hip_bf16.h>

#define DIM     256
#define NNODES  256
#define NPROB   64
#define NGRAPH  16
#define VDIM    300

typedef float v2f __attribute__((ext_vector_type(2)));
typedef float v8f __attribute__((ext_vector_type(8)));
typedef int   vi4 __attribute__((__vector_size__(16)));

#if defined(__HIP_DEVICE_COMPILE__)
# if __has_builtin(__builtin_amdgcn_wmma_f32_16x16x4_f32)
#  define HAVE_WMMA_F32 1
# endif
# if __has_builtin(__builtin_amdgcn_global_load_async_to_lds_b128) && \
     __has_builtin(__builtin_amdgcn_s_wait_asynccnt)
#  define HAVE_ASYNC_LDS 1
# endif
typedef __bf16 v16bf __attribute__((ext_vector_type(16)));
#endif

#define AS_GLOBAL __attribute__((address_space(1)))
#define AS_LOCAL  __attribute__((address_space(3)))

// ---------------------------------------------------------------------------
// Kernel A: base[m][n] = sum_k data_vecs[data[m]][k] * data_w[k][n] + data_b[n]
// 256x300 @ 300x256, WMMA tiles: 16x16 output per wave, 256 tiles total.
// ---------------------------------------------------------------------------
__global__ __launch_bounds__(256) void base_gemm_kernel(
    const int* __restrict__ data,
    const float* __restrict__ data_vecs,
    const float* __restrict__ data_w,
    const float* __restrict__ data_b,
    float* __restrict__ base)
{
    const int lane = threadIdx.x & 31;
    const int wave = threadIdx.x >> 5;
    const int tile = blockIdx.x * 8 + wave;      // 0..255
    const int tm = tile >> 4, tn = tile & 15;
    const int hi = lane >> 4, lm = lane & 15;
    const int n0 = tn * 16;

    const int tok = data[tm * 16 + lm];
    const float* arow = data_vecs + (size_t)tok * VDIM;

    v8f acc = {};
#if defined(__HIP_DEVICE_COMPILE__)
# if HAVE_WMMA_F32
    for (int k = 0; k < VDIM; k += 4) {
        const int ka = k + hi * 2;
        v2f a, b;
        a.x = arow[ka];
        a.y = arow[ka + 1];
        b.x = data_w[(size_t)ka * DIM + n0 + lm];
        b.y = data_w[(size_t)(ka + 1) * DIM + n0 + lm];
        acc = __builtin_amdgcn_wmma_f32_16x16x4_f32(
            false, a, false, b, (short)0, acc, false, false);
    }
# else
    for (int k = 0; k < 320; k += 32) {          // pad K 300 -> 320 with zeros
        v16bf a, b;
        for (int e2 = 0; e2 < 16; ++e2) {
            const int vg = e2 >> 1, pos = e2 & 1;
            const int kk = k + ((vg & 3) << 1) + ((vg >> 2) << 4) + hi * 8 + pos;
            a[e2] = (kk < VDIM) ? (__bf16)arow[kk] : (__bf16)0.0f;
            b[e2] = (kk < VDIM) ? (__bf16)data_w[(size_t)kk * DIM + n0 + lm]
                                : (__bf16)0.0f;
        }
        acc = __builtin_amdgcn_wmma_f32_16x16x32_bf16(
            false, a, false, b, (short)0, acc, false, false);
    }
# endif
#endif
    for (int j = 0; j < 8; ++j) {
        const int m = tm * 16 + j + hi * 8;
        const int n = n0 + lm;
        base[m * DIM + n] = acc[j] + data_b[n];
    }
}

// ---------------------------------------------------------------------------
// Kernel B: c_e[e][n] = base[255] @ edge_w[e] + edge_b[e]  (4 matvecs),
//           data_score, and per-problem variant edge selection.
// ---------------------------------------------------------------------------
__global__ __launch_bounds__(256) void prep_kernel(
    const int* __restrict__ edges,
    const float* __restrict__ base,
    const float* __restrict__ edge_w,
    const float* __restrict__ edge_b,
    const float* __restrict__ score_data_w,
    const float* __restrict__ score_data_b,
    float* __restrict__ c_e,          // 4*256
    float* __restrict__ scalars,      // [0] = data_score
    int* __restrict__ edge_sel)       // 64
{
    __shared__ float red[256];
    const int n = threadIdx.x;
    const float* b255 = base + 255 * DIM;

    for (int e = 0; e < 4; ++e) {
        float s = edge_b[e * DIM + n];
        const float* W = edge_w + (size_t)e * DIM * DIM;
        for (int k = 0; k < DIM; ++k)
            s = fmaf(b255[k], W[(size_t)k * DIM + n], s);
        c_e[e * DIM + n] = s;
    }

    red[n] = (b255[n] + score_data_b[n]) * score_data_w[n];
    __syncthreads();
    for (int s = 128; s > 0; s >>= 1) {
        if (n < s) red[n] += red[n + s];
        __syncthreads();
    }
    if (n == 0) scalars[0] = red[0];

    if (n < NPROB) {
        const int correct = edges[255];
        int alts[3]; int idx = 0;
        for (int e = 0; e < 4; ++e) if (e != correct) alts[idx++] = e;
        const int g = n >> 2, v = n & 3;
        edge_sel[n] = (v == 0) ? (g == 0 ? correct : alts[2]) : alts[v - 1];
    }
}

// ---------------------------------------------------------------------------
// Kernel C: emb[p][node][n] = base[node][n] (+ c_e[sel(p)][n] at par_g(255))
// ---------------------------------------------------------------------------
__global__ __launch_bounds__(256) void init_emb_kernel(
    const int* __restrict__ graphs,
    const float* __restrict__ base,
    const float* __restrict__ c_e,
    const int* __restrict__ edge_sel,
    float* __restrict__ emb)
{
    const size_t idx = (size_t)blockIdx.x * 256 + threadIdx.x;  // < 64*65536
    const int p    = (int)(idx >> 16);
    const int node = (int)((idx >> 8) & 255);
    const int n    = (int)(idx & 255);
    float val = base[node * DIM + n];
    const int g = p >> 2;
    const int par255 = 255 + graphs[g * NNODES + 255];
    if (node == par255)
        val += c_e[edge_sel[p] * DIM + n];
    emb[idx] = val;
}

// ---------------------------------------------------------------------------
// Kernel D: persistent single-workgroup tree propagation.
// For i = 254..1:  C(64x256) = Emb_i(64x256) @ edge_w[e_i] + edge_b[e_i],
//                  emb[p][par_g(i)] += C[p].
// Then score_p = dot(emb[p][0] + score_emb_b, score_emb_w) + data_score.
// 16 waves: wave w owns N-tile w (16 cols), iterates the 4 M-tiles.
// A rows staged to LDS via GLOBAL_LOAD_ASYNC_TO_LDS_B128 when available.
// ---------------------------------------------------------------------------
__global__ __launch_bounds__(512) void tree_kernel(
    const int* __restrict__ edges,
    const int* __restrict__ graphs,
    const float* __restrict__ edge_w,
    const float* __restrict__ edge_b,
    const float* __restrict__ score_emb_w,
    const float* __restrict__ score_emb_b,
    const float* __restrict__ scalars,
    float* __restrict__ emb,
    float* __restrict__ out)
{
    __shared__ __align__(16) float As[NPROB][DIM];   // 64 KB staging for A rows
    const int t = threadIdx.x;
    const int lane = t & 31;
    const int wave = t >> 5;           // 0..15
    const int hi = lane >> 4;
    const int lm = lane & 15;
    const int n0 = wave * 16;

    for (int i = 254; i >= 1; --i) {
        const int e = edges[i];
        const float* W    = edge_w + (size_t)e * DIM * DIM;
        const float* bias = edge_b + e * DIM;

        // Stage A rows (emb[p][i][:]) into LDS: 64 KB, b128 transfers.
#if defined(__HIP_DEVICE_COMPILE__) && HAVE_ASYNC_LDS
        for (int q = t; q < NPROB * (DIM / 4); q += 512) {
            const int p = q >> 6, c4 = q & 63;
            __builtin_amdgcn_global_load_async_to_lds_b128(
                (AS_GLOBAL vi4*)((const float4*)(emb + ((size_t)p * NNODES + i) * DIM) + c4),
                (AS_LOCAL vi4*)(&As[p][c4 * 4]), 0, 0);
        }
#else
        for (int q = t; q < NPROB * (DIM / 4); q += 512) {
            const int p = q >> 6, c4 = q & 63;
            ((float4*)(&As[p][0]))[c4] =
                ((const float4*)(emb + ((size_t)p * NNODES + i) * DIM))[c4];
        }
#endif

        // Prefetch next step's weight matrix into cache (global_prefetch_b8),
        // overlapped with the (async) LDS staging above.
        if (i > 1) {
            const char* Wn = (const char*)(edge_w + (size_t)edges[i - 1] * DIM * DIM);
            __builtin_prefetch(Wn + (size_t)t * 512 +   0, 0, 0);
            __builtin_prefetch(Wn + (size_t)t * 512 + 128, 0, 0);
            __builtin_prefetch(Wn + (size_t)t * 512 + 256, 0, 0);
            __builtin_prefetch(Wn + (size_t)t * 512 + 384, 0, 0);
        }

#if defined(__HIP_DEVICE_COMPILE__) && HAVE_ASYNC_LDS
        __builtin_amdgcn_s_wait_asynccnt(0);   // own wave's DMA done
#endif
        __syncthreads();                        // all waves' staging visible

        for (int r = 0; r < 4; ++r) {
            v8f acc = {};
#if defined(__HIP_DEVICE_COMPILE__)
# if HAVE_WMMA_F32
            for (int k = 0; k < DIM; k += 4) {
                const int ka = k + hi * 2;
                v2f a, b;
                a.x = As[r * 16 + lm][ka];
                a.y = As[r * 16 + lm][ka + 1];
                b.x = W[(size_t)ka * DIM + n0 + lm];
                b.y = W[(size_t)(ka + 1) * DIM + n0 + lm];
                acc = __builtin_amdgcn_wmma_f32_16x16x4_f32(
                    false, a, false, b, (short)0, acc, false, false);
            }
# else
            for (int k = 0; k < DIM; k += 32) {
                v16bf a, b;
                for (int e2 = 0; e2 < 16; ++e2) {
                    const int vg = e2 >> 1, pos = e2 & 1;
                    const int kk = k + ((vg & 3) << 1) + ((vg >> 2) << 4) + hi * 8 + pos;
                    a[e2] = (__bf16)As[r * 16 + lm][kk];
                    b[e2] = (__bf16)W[(size_t)kk * DIM + n0 + lm];
                }
                acc = __builtin_amdgcn_wmma_f32_16x16x32_bf16(
                    false, a, false, b, (short)0, acc, false, false);
            }
# endif
#endif
            // Scatter-add into parent rows (unique (p, n) per lane -> no atomics).
            const float bn = bias[n0 + lm];
            for (int j = 0; j < 8; ++j) {
                const int p = r * 16 + j + hi * 8;
                const int g = p >> 2;
                const int par = i + graphs[g * NNODES + i];
                float* dst = emb + ((size_t)p * NNODES + par) * DIM + n0 + lm;
                *dst += acc[j] + bn;
            }
        }
        __threadfence();      // make global scatter visible across both CUs
        __syncthreads();
    }

    // Final scoring: out[p] = dot(emb[p][0] + score_emb_b, score_emb_w) + data_score
    float* red = &As[0][0];   // reuse LDS
    const float dscore = scalars[0];
    {
        const int p = t >> 3, l8 = t & 7;
        float s = 0.0f;
        const float* root = emb + (size_t)p * NNODES * DIM;   // node 0 row
        for (int n = l8; n < DIM; n += 8)
            s = fmaf(root[n] + score_emb_b[n], score_emb_w[n], s);
        red[t] = s;
    }
    __syncthreads();
    if ((t & 7) == 0) {
        float tot = dscore;
        for (int q = 0; q < 8; ++q) tot += red[t + q];
        out[t >> 3] = tot;
    }
}

// ---------------------------------------------------------------------------
extern "C" void kernel_launch(void* const* d_in, const int* in_sizes, int n_in,
                              void* d_out, int out_size, void* d_ws, size_t ws_size,
                              hipStream_t stream)
{
    const int*   data         = (const int*)d_in[0];
    // d_in[1] = types (unused)
    const int*   edges        = (const int*)d_in[2];
    const int*   graphs       = (const int*)d_in[3];
    const float* data_vecs    = (const float*)d_in[4];
    const float* data_w       = (const float*)d_in[5];
    const float* data_b       = (const float*)d_in[6];
    const float* edge_w       = (const float*)d_in[7];
    const float* edge_b       = (const float*)d_in[8];
    const float* score_emb_w  = (const float*)d_in[9];
    const float* score_emb_b  = (const float*)d_in[10];
    const float* score_data_w = (const float*)d_in[11];
    const float* score_data_b = (const float*)d_in[12];
    float* out = (float*)d_out;

    // Workspace layout (floats):
    //   base     @ 0        : 256*256      = 65536
    //   c_e      @ 65536    : 4*256        = 1024
    //   scalars  @ 66560    : 1
    //   edge_sel @ 66564    : 64 ints
    //   emb      @ 66816    : 64*256*256   = 4194304   (16-byte aligned)
    float* ws      = (float*)d_ws;
    float* base    = ws;
    float* c_e     = ws + 65536;
    float* scalars = ws + 66560;
    int*   edge_sel= (int*)(ws + 66564);
    float* emb     = ws + 66816;

    base_gemm_kernel<<<32, 256, 0, stream>>>(data, data_vecs, data_w, data_b, base);
    prep_kernel<<<1, 256, 0, stream>>>(edges, base, edge_w, edge_b,
                                       score_data_w, score_data_b,
                                       c_e, scalars, edge_sel);
    init_emb_kernel<<<NPROB * NNODES * DIM / 256, 256, 0, stream>>>(
        graphs, base, c_e, edge_sel, emb);
    tree_kernel<<<1, 512, 0, stream>>>(edges, graphs, edge_w, edge_b,
                                       score_emb_w, score_emb_b,
                                       scalars, emb, out);
}
